// MoELayer_22823456211127
// MI455X (gfx1250) — compile-verified, matrix operand order checked by reading, not run on hardware
//
#include <hip/hip_runtime.h>
#include <hip/hip_bf16.h>

#define TT 8192   // B*S tokens
#define HH 1024
#define FF 4096
#define EE 8

typedef __attribute__((ext_vector_type(16))) __bf16 v16bf;
typedef __attribute__((ext_vector_type(8)))  __bf16 v8bf;
typedef __attribute__((ext_vector_type(8)))  float  v8f;

#if defined(__AMDGCN__) && __has_builtin(__builtin_amdgcn_global_load_async_to_lds_b128)
#define HAVE_ASYNC_LDS 1
#else
#define HAVE_ASYNC_LDS 0
#endif

__device__ inline v8f vzero8() {
  v8f v;
  for (int i = 0; i < 8; ++i) v[i] = 0.0f;
  return v;
}

#if HAVE_ASYNC_LDS
// Exact pointee type per clang's diagnostic: int __attribute__((vector_size(16)))
typedef int v4i_vs __attribute__((__vector_size__(4 * sizeof(int))));
typedef __attribute__((address_space(1))) v4i_vs* g_v4i_ptr;  // global
typedef __attribute__((address_space(3))) v4i_vs* l_v4i_ptr;  // LDS

// 16B per-lane async copy global -> LDS (GLOBAL_LOAD_ASYNC_TO_LDS_B128, ASYNCcnt)
__device__ inline void async_cp16(void* dst_lds, const void* src_global) {
  __builtin_amdgcn_global_load_async_to_lds_b128(
      (g_v4i_ptr)src_global, (l_v4i_ptr)dst_lds, 0, 0);
}
__device__ inline void wait_async0() {
#if __has_builtin(__builtin_amdgcn_s_wait_asynccnt)
  __builtin_amdgcn_s_wait_asynccnt(0);
#else
  asm volatile("s_wait_asynccnt 0" ::: "memory");
#endif
}
#endif

// Load a 16x32 bf16 fragment (A-layout per CDNA5 ISA 7.12.2) from LDS.
// base points at row m=0 of the 16-row subtile; rows are 48 elements apart.
__device__ inline v16bf ld_frag(const __bf16* base, int lane) {
  const int m = lane & 15, half = lane >> 4;
  const __bf16* p = base + m * 48;
  union { v16bf v; v8bf h[2]; } u;
  u.h[0] = *(const v8bf*)(p + half * 8);        // K = half*8 .. half*8+7
  u.h[1] = *(const v8bf*)(p + 16 + half * 8);   // K = 16+half*8 .. +7
  return u.v;
}

// ---------------------------------------------------------------- gating ----
__global__ __launch_bounds__(256) void k_gate(
    const float* __restrict__ x, const float* __restrict__ gn_g,
    const float* __restrict__ gn_b, const float* __restrict__ gate_w,
    const float* __restrict__ gate_b, __bf16* __restrict__ xb,
    int* __restrict__ counts, int* __restrict__ list,
    float* __restrict__ probs, int* __restrict__ pick) {
  __shared__ float sm[512];
  __shared__ float g8[256 * 8];
  const int t = blockIdx.x, tid = threadIdx.x;
  const float4 v = ((const float4*)(x + (size_t)t * HH))[tid];

  // bf16 copy of x for the WMMA GEMMs
  __bf16* xr = xb + (size_t)t * HH + tid * 4;
  xr[0] = (__bf16)v.x; xr[1] = (__bf16)v.y; xr[2] = (__bf16)v.z; xr[3] = (__bf16)v.w;

  // mean / var over H
  float s1 = v.x + v.y + v.z + v.w;
  float s2 = v.x * v.x + v.y * v.y + v.z * v.z + v.w * v.w;
  sm[tid] = s1; sm[256 + tid] = s2;
  __syncthreads();
  for (int st = 128; st > 0; st >>= 1) {
    if (tid < st) { sm[tid] += sm[tid + st]; sm[256 + tid] += sm[256 + tid + st]; }
    __syncthreads();
  }
  const float mean = sm[0] * (1.0f / HH);
  const float var  = sm[256] * (1.0f / HH) - mean * mean;
  const float rstd = rsqrtf(var + 1e-5f);

  // partial logits
  float pl[EE];
  for (int e = 0; e < EE; ++e) pl[e] = 0.0f;
  const float xv[4] = {v.x, v.y, v.z, v.w};
  for (int j = 0; j < 4; ++j) {
    const int i = tid * 4 + j;
    const float xn = (xv[j] - mean) * rstd * gn_g[i] + gn_b[i];
    for (int e = 0; e < EE; ++e) pl[e] += xn * gate_w[(size_t)i * EE + e];
  }
  for (int e = 0; e < EE; ++e) g8[tid * EE + e] = pl[e];
  __syncthreads();
  for (int st = 128; st > 0; st >>= 1) {
    if (tid < st)
      for (int e = 0; e < EE; ++e) g8[tid * EE + e] += g8[(tid + st) * EE + e];
    __syncthreads();
  }

  if (tid == 0) {
    float l[EE], mx = -1e30f;
    for (int e = 0; e < EE; ++e) { l[e] = g8[e] + gate_b[e]; mx = fmaxf(mx, l[e]); }
    float den = 0.0f;
    for (int e = 0; e < EE; ++e) { l[e] = __expf(l[e] - mx); den += l[e]; }
    int i0 = 0; float p0 = -1.0f;
    for (int e = 0; e < EE; ++e) if (l[e] > p0) { p0 = l[e]; i0 = e; }
    int i1 = -1; float p1 = -1.0f;
    for (int e = 0; e < EE; ++e) if (e != i0 && l[e] > p1) { p1 = l[e]; i1 = e; }
    p0 /= den; p1 /= den;
    const float rn = 1.0f / (p0 + p1 + 1e-9f);
    pick[t * 2 + 0] = i0; pick[t * 2 + 1] = i1;
    probs[t * 2 + 0] = p0 * rn; probs[t * 2 + 1] = p1 * rn;
    int pos0 = atomicAdd(counts + i0, 1);
    list[i0 * TT + pos0] = t * 2 + 0;
    int pos1 = atomicAdd(counts + i1, 1);
    list[i1 * TT + pos1] = t * 2 + 1;
  }
}

// ----------------------------------------------- weight convert+transpose ----
// in: [E][R][C] f32  ->  out: [E][C][R] bf16
__global__ void k_transpose(const float* __restrict__ in, __bf16* __restrict__ out,
                            int R, int C) {
  __shared__ __bf16 tile[32][33];
  const int e = blockIdx.z;
  const int c0 = blockIdx.x * 32, r0 = blockIdx.y * 32;
  const float* src = in + ((size_t)e * R + r0) * C + c0;
  for (int i = threadIdx.y; i < 32; i += 8)
    tile[i][threadIdx.x] = (__bf16)src[(size_t)i * C + threadIdx.x];
  __syncthreads();
  __bf16* dst = out + ((size_t)e * C + c0) * R + r0;
  for (int i = threadIdx.y; i < 32; i += 8)
    dst[(size_t)i * R + threadIdx.x] = tile[threadIdx.x][i];
}

// ---------------------------------------------------------------- GEMM core --
// Double-buffered, gathered-A, WMMA-bf16 GEMM; async global->LDS staging when
// the gfx1250 async builtins are available, register-staged ping-pong otherwise.

// GEMM1: h[slot][F] = gelu( x[tok] @ W1[e] + b1[e] ), gathered by expert list
__global__ __launch_bounds__(256) void k_gemm1(
    const __bf16* __restrict__ xb, const __bf16* __restrict__ w1t,
    const float* __restrict__ b1, const int* __restrict__ counts,
    const int* __restrict__ list, __bf16* __restrict__ h) {
  __shared__ __align__(16) __bf16 Al[2][64 * 48];
  __shared__ __align__(16) __bf16 Bl[2][128 * 48];
  __shared__ int sl[64];

  const int e = blockIdx.z;
  const int cnt = counts[e];
  const int mbase = blockIdx.y * 64;
  if (mbase >= cnt) return;
  const int nbase = blockIdx.x * 128;
  const int tid = threadIdx.x;
  const int lane = tid & 31, w = tid >> 5;
  const int wm = w >> 2, wn = w & 3;

  if (tid < 64) {
    const int p = mbase + tid;
    sl[tid] = (p < cnt) ? list[e * TT + p] : -1;
  }
  __syncthreads();

  v8f c[2][2];
  for (int i = 0; i < 2; ++i)
    for (int j = 0; j < 2; ++j) c[i][j] = vzero8();

  const int arow = tid >> 2, aseg = tid & 3;
  const int s_a = sl[arow];
  const __bf16* asrc = xb + (size_t)((s_a < 0) ? 0 : (s_a >> 1)) * HH + aseg * 8;
  __bf16* adst = &Al[0][arow * 48 + aseg * 8];         // +buf*64*48
  const int bn0 = tid >> 1, bseg0 = (tid & 1) * 2;     // two 16B segs per thread
  const __bf16* bsrc = w1t + ((size_t)e * FF + nbase + bn0) * HH + bseg0 * 8;
  __bf16* bdst = &Bl[0][bn0 * 48 + bseg0 * 8];

  const int NK = HH / 32;
#if HAVE_ASYNC_LDS
  // prologue: async-stage tile 0
  async_cp16(adst, asrc);
  async_cp16(bdst, bsrc);
  async_cp16(bdst + 8, bsrc + 8);
  wait_async0();
  __syncthreads();
  for (int it = 0; it < NK; ++it) {
    const int kn = (it + 1) * 32;
    const int nb = (it + 1) & 1;
    if (kn < HH) {  // stage next tile while computing current
      async_cp16(adst + nb * (64 * 48), asrc + kn);
      async_cp16(bdst + nb * (128 * 48), bsrc + kn);
      async_cp16(bdst + nb * (128 * 48) + 8, bsrc + kn + 8);
    }
    const __bf16* Ab = Al[it & 1];
    const __bf16* Bb = Bl[it & 1];
    v16bf a[2], b[2];
    a[0] = ld_frag(Ab + (wm * 32) * 48, lane);
    a[1] = ld_frag(Ab + (wm * 32 + 16) * 48, lane);
    b[0] = ld_frag(Bb + (wn * 32) * 48, lane);
    b[1] = ld_frag(Bb + (wn * 32 + 16) * 48, lane);
#pragma unroll
    for (int i = 0; i < 2; ++i)
#pragma unroll
      for (int j = 0; j < 2; ++j)
        c[i][j] = __builtin_amdgcn_wmma_f32_16x16x32_bf16(
            false, a[i], false, b[j], (short)0, c[i][j], false, false);
    wait_async0();
    __syncthreads();
  }
#else
  // prologue: register-stage tile 0
  v8bf ra = *(const v8bf*)(asrc);
  v8bf rb0 = *(const v8bf*)(bsrc);
  v8bf rb1 = *(const v8bf*)(bsrc + 8);
  *(v8bf*)adst = ra;
  *(v8bf*)bdst = rb0;
  *(v8bf*)(bdst + 8) = rb1;
  __syncthreads();
  for (int it = 0; it < NK; ++it) {
    const int kn = (it + 1) * 32;
    const int nb = (it + 1) & 1;
    const bool more = kn < HH;
    if (more) {  // issue next-tile global loads before the math
      ra = *(const v8bf*)(asrc + kn);
      rb0 = *(const v8bf*)(bsrc + kn);
      rb1 = *(const v8bf*)(bsrc + kn + 8);
    }
    const __bf16* Ab = Al[it & 1];
    const __bf16* Bb = Bl[it & 1];
    v16bf a[2], b[2];
    a[0] = ld_frag(Ab + (wm * 32) * 48, lane);
    a[1] = ld_frag(Ab + (wm * 32 + 16) * 48, lane);
    b[0] = ld_frag(Bb + (wn * 32) * 48, lane);
    b[1] = ld_frag(Bb + (wn * 32 + 16) * 48, lane);
#pragma unroll
    for (int i = 0; i < 2; ++i)
#pragma unroll
      for (int j = 0; j < 2; ++j)
        c[i][j] = __builtin_amdgcn_wmma_f32_16x16x32_bf16(
            false, a[i], false, b[j], (short)0, c[i][j], false, false);
    if (more) {
      *(v8bf*)(adst + nb * (64 * 48)) = ra;
      *(v8bf*)(bdst + nb * (128 * 48)) = rb0;
      *(v8bf*)(bdst + nb * (128 * 48) + 8) = rb1;
    }
    __syncthreads();
  }
#endif

  const int half = lane >> 4, nn = lane & 15;
#pragma unroll
  for (int i = 0; i < 2; ++i)
#pragma unroll
    for (int j = 0; j < 2; ++j)
#pragma unroll
      for (int r = 0; r < 8; ++r) {
        const int ml = wm * 32 + i * 16 + r + 8 * half;
        const int nl = wn * 32 + j * 16 + nn;
        const int s = sl[ml];
        if (s < 0) continue;
        float v = c[i][j][r] + b1[e * FF + nbase + nl];
        v = 0.5f * v * (1.0f + erff(v * 0.70710678118f));  // exact gelu
        h[(size_t)s * FF + nbase + nl] = (__bf16)v;
      }
}

// GEMM2: out2[slot][H] = h[slot] @ W2[e] + b2[e]
__global__ __launch_bounds__(256) void k_gemm2(
    const __bf16* __restrict__ hin, const __bf16* __restrict__ w2t,
    const float* __restrict__ b2, const int* __restrict__ counts,
    const int* __restrict__ list, float* __restrict__ out2) {
  __shared__ __align__(16) __bf16 Al[2][64 * 48];
  __shared__ __align__(16) __bf16 Bl[2][128 * 48];
  __shared__ int sl[64];

  const int e = blockIdx.z;
  const int cnt = counts[e];
  const int mbase = blockIdx.y * 64;
  if (mbase >= cnt) return;
  const int nbase = blockIdx.x * 128;
  const int tid = threadIdx.x;
  const int lane = tid & 31, w = tid >> 5;
  const int wm = w >> 2, wn = w & 3;

  if (tid < 64) {
    const int p = mbase + tid;
    sl[tid] = (p < cnt) ? list[e * TT + p] : -1;
  }
  __syncthreads();

  v8f c[2][2];
  for (int i = 0; i < 2; ++i)
    for (int j = 0; j < 2; ++j) c[i][j] = vzero8();

  const int arow = tid >> 2, aseg = tid & 3;
  const int s_a = sl[arow];
  const __bf16* asrc = hin + (size_t)((s_a < 0) ? 0 : s_a) * FF + aseg * 8;
  __bf16* adst = &Al[0][arow * 48 + aseg * 8];
  const int bn0 = tid >> 1, bseg0 = (tid & 1) * 2;
  const __bf16* bsrc = w2t + ((size_t)e * HH + nbase + bn0) * FF + bseg0 * 8;
  __bf16* bdst = &Bl[0][bn0 * 48 + bseg0 * 8];

  const int NK = FF / 32;
#if HAVE_ASYNC_LDS
  async_cp16(adst, asrc);
  async_cp16(bdst, bsrc);
  async_cp16(bdst + 8, bsrc + 8);
  wait_async0();
  __syncthreads();
  for (int it = 0; it < NK; ++it) {
    const int kn = (it + 1) * 32;
    const int nb = (it + 1) & 1;
    if (kn < FF) {
      async_cp16(adst + nb * (64 * 48), asrc + kn);
      async_cp16(bdst + nb * (128 * 48), bsrc + kn);
      async_cp16(bdst + nb * (128 * 48) + 8, bsrc + kn + 8);
    }
    const __bf16* Ab = Al[it & 1];
    const __bf16* Bb = Bl[it & 1];
    v16bf a[2], b[2];
    a[0] = ld_frag(Ab + (wm * 32) * 48, lane);
    a[1] = ld_frag(Ab + (wm * 32 + 16) * 48, lane);
    b[0] = ld_frag(Bb + (wn * 32) * 48, lane);
    b[1] = ld_frag(Bb + (wn * 32 + 16) * 48, lane);
#pragma unroll
    for (int i = 0; i < 2; ++i)
#pragma unroll
      for (int j = 0; j < 2; ++j)
        c[i][j] = __builtin_amdgcn_wmma_f32_16x16x32_bf16(
            false, a[i], false, b[j], (short)0, c[i][j], false, false);
    wait_async0();
    __syncthreads();
  }
#else
  v8bf ra = *(const v8bf*)(asrc);
  v8bf rb0 = *(const v8bf*)(bsrc);
  v8bf rb1 = *(const v8bf*)(bsrc + 8);
  *(v8bf*)adst = ra;
  *(v8bf*)bdst = rb0;
  *(v8bf*)(bdst + 8) = rb1;
  __syncthreads();
  for (int it = 0; it < NK; ++it) {
    const int kn = (it + 1) * 32;
    const int nb = (it + 1) & 1;
    const bool more = kn < FF;
    if (more) {
      ra = *(const v8bf*)(asrc + kn);
      rb0 = *(const v8bf*)(bsrc + kn);
      rb1 = *(const v8bf*)(bsrc + kn + 8);
    }
    const __bf16* Ab = Al[it & 1];
    const __bf16* Bb = Bl[it & 1];
    v16bf a[2], b[2];
    a[0] = ld_frag(Ab + (wm * 32) * 48, lane);
    a[1] = ld_frag(Ab + (wm * 32 + 16) * 48, lane);
    b[0] = ld_frag(Bb + (wn * 32) * 48, lane);
    b[1] = ld_frag(Bb + (wn * 32 + 16) * 48, lane);
#pragma unroll
    for (int i = 0; i < 2; ++i)
#pragma unroll
      for (int j = 0; j < 2; ++j)
        c[i][j] = __builtin_amdgcn_wmma_f32_16x16x32_bf16(
            false, a[i], false, b[j], (short)0, c[i][j], false, false);
    if (more) {
      *(v8bf*)(adst + nb * (64 * 48)) = ra;
      *(v8bf*)(bdst + nb * (128 * 48)) = rb0;
      *(v8bf*)(bdst + nb * (128 * 48) + 8) = rb1;
    }
    __syncthreads();
  }
#endif

  const int half = lane >> 4, nn = lane & 15;
#pragma unroll
  for (int i = 0; i < 2; ++i)
#pragma unroll
    for (int j = 0; j < 2; ++j)
#pragma unroll
      for (int r = 0; r < 8; ++r) {
        const int ml = wm * 32 + i * 16 + r + 8 * half;
        const int nl = wn * 32 + j * 16 + nn;
        const int s = sl[ml];
        if (s < 0) continue;
        out2[(size_t)s * HH + nbase + nl] = c[i][j][r] + b2[e * HH + nbase + nl];
      }
}

// ---------------------------------------------------------------- combine ----
__global__ __launch_bounds__(256) void k_combine(
    const float* __restrict__ x, const float* __restrict__ out2,
    const float* __restrict__ probs, const int* __restrict__ pick,
    const float* __restrict__ ln_g, const float* __restrict__ ln_b,
    float* __restrict__ y) {
  __shared__ float sm[512];
  const int t = blockIdx.x, tid = threadIdx.x;
  const float4 xr = ((const float4*)(x + (size_t)t * HH))[tid];
  float acc[4] = {0, 0, 0, 0};

  for (int k = 0; k < 2; ++k) {
    const int e = pick[t * 2 + k];
    const float p = probs[t * 2 + k];
    const float4 o = ((const float4*)(out2 + (size_t)(t * 2 + k) * HH))[tid];
    const float r[4] = {o.x + xr.x, o.y + xr.y, o.z + xr.z, o.w + xr.w};
    const float s1 = r[0] + r[1] + r[2] + r[3];
    const float s2 = r[0] * r[0] + r[1] * r[1] + r[2] * r[2] + r[3] * r[3];
    __syncthreads();  // protect sm reuse across k iterations
    sm[tid] = s1; sm[256 + tid] = s2;
    __syncthreads();
    for (int st = 128; st > 0; st >>= 1) {
      if (tid < st) { sm[tid] += sm[tid + st]; sm[256 + tid] += sm[256 + tid + st]; }
      __syncthreads();
    }
    const float mean = sm[0] * (1.0f / HH);
    const float var  = sm[256] * (1.0f / HH) - mean * mean;
    const float rstd = rsqrtf(var + 1e-5f);
    for (int j = 0; j < 4; ++j) {
      const int i = tid * 4 + j;
      acc[j] += p * ((r[j] - mean) * rstd * ln_g[(size_t)e * HH + i] + ln_b[(size_t)e * HH + i]);
    }
  }
  float4 yo = {acc[0], acc[1], acc[2], acc[3]};
  ((float4*)(y + (size_t)t * HH))[tid] = yo;
}

// ------------------------------------------------------------------ launch ---
extern "C" void kernel_launch(void* const* d_in, const int* in_sizes, int n_in,
                              void* d_out, int out_size, void* d_ws, size_t ws_size,
                              hipStream_t stream) {
  const float* x      = (const float*)d_in[0];
  const float* W1     = (const float*)d_in[1];
  const float* b1     = (const float*)d_in[2];
  const float* W2     = (const float*)d_in[3];
  const float* b2     = (const float*)d_in[4];
  const float* ln_g   = (const float*)d_in[5];
  const float* ln_b   = (const float*)d_in[6];
  const float* gn_g   = (const float*)d_in[7];
  const float* gn_b   = (const float*)d_in[8];
  const float* gate_w = (const float*)d_in[9];
  const float* gate_b = (const float*)d_in[10];
  float* y = (float*)d_out;

  char* w = (char*)d_ws;
  size_t off = 0;
  auto take = [&](size_t b) -> void* {
    void* p = w + off;
    off = (off + b + 255) & ~(size_t)255;
    return p;
  };
  __bf16* xb   = (__bf16*)take((size_t)TT * HH * 2);
  __bf16* w1t  = (__bf16*)take((size_t)EE * FF * HH * 2);   // [E][F][H]
  __bf16* w2t  = (__bf16*)take((size_t)EE * HH * FF * 2);   // [E][H][F]
  __bf16* hbuf = (__bf16*)take((size_t)2 * TT * FF * 2);    // [2T][F]
  float*  out2 = (float*) take((size_t)2 * TT * HH * 4);    // [2T][H]
  int*    cnts = (int*)   take((size_t)EE * 4);
  int*    list = (int*)   take((size_t)EE * TT * 4);
  float*  prbs = (float*) take((size_t)TT * 2 * 4);
  int*    pick = (int*)   take((size_t)TT * 2 * 4);

  (void)hipMemsetAsync(cnts, 0, EE * sizeof(int), stream);

  k_gate<<<TT, 256, 0, stream>>>(x, gn_g, gn_b, gate_w, gate_b, xb, cnts, list, prbs, pick);
  k_transpose<<<dim3(FF / 32, HH / 32, EE), dim3(32, 8), 0, stream>>>(W1, w1t, HH, FF);
  k_transpose<<<dim3(HH / 32, FF / 32, EE), dim3(32, 8), 0, stream>>>(W2, w2t, FF, HH);
  k_gemm1<<<dim3(FF / 128, TT / 64, EE), 256, 0, stream>>>(xb, w1t, b1, cnts, list, hbuf);
  k_gemm2<<<dim3(HH / 128, TT / 64, EE), 256, 0, stream>>>(hbuf, w2t, b2, cnts, list, out2);
  k_combine<<<TT, 256, 0, stream>>>(x, out2, prbs, pick, ln_g, ln_b, y);
}
